// Transformer_66700842107476
// MI455X (gfx1250) — matrix operand order repaired and checked by reading, not compiled, not run on hardware
//
#include <hip/hip_runtime.h>

typedef __attribute__((ext_vector_type(16))) _Float16 v16h;
typedef __attribute__((ext_vector_type(8)))  float    v8f;

#define TPB  256
#define BM   128
#define BN   64
#define BK   32
#define LPAD 40   // halves per LDS row: 32 data + 8 pad (16B) -> matches TDM pad cfg

// ---------------------------------------------------------------------------
// Tensor Data Mover: 2D tile (tileX x tileY elements, 2B each) global -> LDS,
// LDS rows padded 16B per 64B (pad_interval=16DW, pad_amount=4DW) == LPAD=40.
// ---------------------------------------------------------------------------
#if __has_builtin(__builtin_amdgcn_tensor_load_to_lds)
#define HAVE_TDM 1
typedef unsigned int u32x4 __attribute__((ext_vector_type(4)));
typedef int          i32x4 __attribute__((ext_vector_type(4)));
typedef int          i32x8 __attribute__((ext_vector_type(8)));

__device__ __forceinline__ void tdm_load_2d(unsigned ldsOff, const _Float16* g,
                                            int tensX, int tensY, int strideX,
                                            int tileX, int tileY) {
  unsigned long long ga = (unsigned long long)(size_t)g;
  u32x4 g0;
  g0[0] = 1u;                                                   // count=1, user D#
  g0[1] = ldsOff;                                               // lds_addr (bytes)
  g0[2] = (unsigned)ga;                                         // global_addr[31:0]
  g0[3] = (unsigned)((ga >> 32) & 0x01ffffffull) | (2u << 30);  // addr[56:32] | type=2
  i32x8 g1;
  g1[0] = (1 << 16)      // data_size = 1 -> 2 bytes
        | (1 << 20)      // pad_enable
        | (3 << 22)      // pad_interval: 16 DWORDs (64B) between pads
        | (3 << 25);     // pad_amount:   4 DWORDs (16B)
  g1[1] = (tensX & 0xffff) << 16;                               // tensor_dim0[15:0]
  g1[2] = (((unsigned)tensX >> 16) & 0xffff) | ((tensY & 0xffff) << 16);
  g1[3] = (((unsigned)tensY >> 16) & 0xffff) | ((tileX & 0xffff) << 16);
  g1[4] = tileY & 0xffff;                                       // tile_dim1 (tile_dim2=0)
  g1[5] = strideX;                                              // tensor_dim0_stride[31:0]
  g1[6] = 0;
  g1[7] = 0;
  i32x4 z4 = {0, 0, 0, 0};
#if __clang_major__ >= 23
  i32x8 z8 = {0, 0, 0, 0, 0, 0, 0, 0};
  __builtin_amdgcn_tensor_load_to_lds(g0, g1, z4, z4, z8, 0);
#else
  __builtin_amdgcn_tensor_load_to_lds(g0, g1, z4, z4, 0);
#endif
}
#endif

// ---- WMMA fragment loaders (layouts per CDNA5 ISA 7.12.2) -----------------
__device__ __forceinline__ v16h frag_a(const _Float16* lA, int m, int h) {
  union { v16h v; unsigned u[8]; } f;
  const unsigned* p0 = (const unsigned*)(lA + m * LPAD + h * 8);
  const unsigned* p1 = (const unsigned*)(lA + m * LPAD + 16 + h * 8);
#pragma unroll
  for (int i = 0; i < 4; ++i) { f.u[i] = p0[i]; f.u[4 + i] = p1[i]; }
  return f.v;
}
__device__ __forceinline__ v16h frag_b(const _Float16* lBt, int n, int h) {
  union { v16h v; unsigned u[8]; } f;
  const unsigned* p0 = (const unsigned*)(lBt + n * LPAD + h * 8);
  const unsigned* p1 = (const unsigned*)(lBt + n * LPAD + 16 + h * 8);
#pragma unroll
  for (int i = 0; i < 4; ++i) { f.u[i] = p0[i]; f.u[4 + i] = p1[i]; }
  return f.v;
}
__device__ __forceinline__ void store_tile(float* C, int ldc, v8f acc, int row0, int col,
                                           const float* bias, float alpha, int relu) {
  float bv = bias ? bias[col] : 0.f;
#pragma unroll
  for (int r = 0; r < 8; ++r) {
    float v = alpha * acc[r] + bv;
    if (relu) v = fmaxf(v, 0.f);
    C[(long long)(row0 + r) * ldc + col] = v;
  }
}

// ---------------------------------------------------------------------------
// Batched GEMM: C[z] = alpha * A[z] * Bt[z]^T + bias (+relu)
// A:  f16 [M][lda] row-major.  Bt: f16 [N][ldb] row-major (i.e. B pre-transposed).
// Batch z = (b = z/innerH, h = z%innerH) with separate strides.
// Feed: TDM double-buffered LDS tiles (fallback: manual f16 copies).
// ---------------------------------------------------------------------------
__global__ __launch_bounds__(TPB) void gemm_f16_wmma(
    const _Float16* __restrict__ A, long long sAb, long long sAh, int lda,
    const _Float16* __restrict__ Bt, long long sBb, long long sBh, int ldb,
    float* __restrict__ C, long long sCb, long long sCh, int ldc,
    const float* __restrict__ bias,
    int M, int N, int K, int innerH, float alpha, int relu)
{
  __shared__ _Float16 sm[2 * BM * LPAD + 2 * BN * LPAD];
  const int aHalf = BM * LPAD;          // halves per A buffer
  const int bBase = 2 * BM * LPAD;      // halves offset of B buffers
  const int bHalf = BN * LPAD;          // halves per B buffer

  const int z  = blockIdx.z;
  const int bb = z / innerH, hh = z % innerH;
  const _Float16* Ab = A  + bb * sAb + hh * sAh;
  const _Float16* Bb = Bt + bb * sBb + hh * sBh;
  C += bb * sCb + hh * sCh;

  const int m0 = blockIdx.y * BM;
  const int n0 = blockIdx.x * BN;
  const int tid  = threadIdx.x;
  const int lane = tid & 31, wave = tid >> 5;
  const int wm = wave >> 1, wn = wave & 1;    // 4x2 waves -> 32x32 per wave
  const int ml = lane & 15, h = lane >> 4;

  v8f acc00 = {}, acc01 = {}, acc10 = {}, acc11 = {};
  const int nsteps = K / BK;

#if defined(HAVE_TDM)
  const unsigned smBase = (unsigned)(size_t)sm;   // LDS byte offset = addr[31:0]
  if (tid < 32) {  // one wave drives the TDM
    tdm_load_2d(smBase,                          Ab + (long long)m0 * lda, K, M, lda, BK, BM);
    tdm_load_2d(smBase + (unsigned)(bBase * 2),  Bb + (long long)n0 * ldb, K, N, ldb, BK, BN);
  }
#endif

  for (int s = 0; s < nsteps; ++s) {
    const int cur = s & 1;
    const int k0 = s * BK;
    _Float16* aC = sm + cur * aHalf;
    _Float16* bC = sm + bBase + cur * bHalf;
#if defined(HAVE_TDM)
    const int nxt = cur ^ 1;
    const bool more = (s + 1 < nsteps);
    if (tid < 32) {
      if (more) {   // prefetch next tile while current computes
        unsigned aOffN = smBase + (unsigned)(nxt * aHalf * 2);
        unsigned bOffN = smBase + (unsigned)((bBase + nxt * bHalf) * 2);
        tdm_load_2d(aOffN, Ab + (long long)m0 * lda + k0 + BK, K, M, lda, BK, BM);
        tdm_load_2d(bOffN, Bb + (long long)n0 * ldb + k0 + BK, K, N, ldb, BK, BN);
      }
      if (more) __builtin_amdgcn_s_wait_tensorcnt(2);  // current tile's 2 TDMs done
      else      __builtin_amdgcn_s_wait_tensorcnt(0);
    }
    __syncthreads();
#else
    {
      const int arow = tid >> 1, akh = (tid & 1) * 16;
      const _Float16* src = Ab + (long long)(m0 + arow) * lda + k0 + akh;
      _Float16* dst = aC + arow * LPAD + akh;
#pragma unroll
      for (int i = 0; i < 16; ++i) dst[i] = src[i];
      const int brow = tid >> 2, bkq = (tid & 3) * 8;
      const _Float16* srcb = Bb + (long long)(n0 + brow) * ldb + k0 + bkq;
      _Float16* dstb = bC + brow * LPAD + bkq;
#pragma unroll
      for (int i = 0; i < 8; ++i) dstb[i] = srcb[i];
    }
    __syncthreads();
#endif
    v16h a0 = frag_a(aC, wm * 32 + ml,      h);
    v16h a1 = frag_a(aC, wm * 32 + 16 + ml, h);
    v16h b0 = frag_b(bC, wn * 32 + ml,      h);
    v16h b1 = frag_b(bC, wn * 32 + 16 + ml, h);
    acc00 = __builtin_amdgcn_wmma_f32_16x16x32_f16(false, a0, false, b0, (short)0, acc00, false, false);
    acc01 = __builtin_amdgcn_wmma_f32_16x16x32_f16(false, a0, false, b1, (short)0, acc01, false, false);
    acc10 = __builtin_amdgcn_wmma_f32_16x16x32_f16(false, a1, false, b0, (short)0, acc10, false, false);
    acc11 = __builtin_amdgcn_wmma_f32_16x16x32_f16(false, a1, false, b1, (short)0, acc11, false, false);
    __syncthreads();   // all waves done reading before that buffer is refilled
  }
  const int rb = m0 + wm * 32 + 8 * h;
  const int cb = n0 + wn * 32 + ml;
  store_tile(C, ldc, acc00, rb,      cb,      bias, alpha, relu);
  store_tile(C, ldc, acc01, rb,      cb + 16, bias, alpha, relu);
  store_tile(C, ldc, acc10, rb + 16, cb,      bias, alpha, relu);
  store_tile(C, ldc, acc11, rb + 16, cb + 16, bias, alpha, relu);
}

// ---- f32 -> f16 flat convert ----------------------------------------------
__global__ __launch_bounds__(256) void cvt_f16_kernel(const float* __restrict__ in,
                                                      _Float16* __restrict__ out, long long n) {
  long long i = (long long)blockIdx.x * 256 + threadIdx.x;
  const long long stride = (long long)gridDim.x * 256;
  for (; i < n; i += stride) out[i] = (_Float16)in[i];
}

// ---- f32 [K][N] -> f16 [N][K] tiled transpose-convert ---------------------
__global__ __launch_bounds__(256) void tpose_f16_kernel(
    const float* __restrict__ in, long long sIn, int ldn,
    _Float16* __restrict__ out, long long sOut, int K, int N) {
  __shared__ float t[32][33];
  in  += (long long)blockIdx.z * sIn;
  out += (long long)blockIdx.z * sOut;
  const int n0 = blockIdx.x * 32, k0 = blockIdx.y * 32;
  const int tx = threadIdx.x & 31, ty = threadIdx.x >> 5;  // 32 x 8
#pragma unroll
  for (int j = 0; j < 4; ++j) {
    int kk = ty + j * 8;
    t[kk][tx] = in[(long long)(k0 + kk) * ldn + n0 + tx];
  }
  __syncthreads();
#pragma unroll
  for (int j = 0; j < 4; ++j) {
    int nn = ty + j * 8;
    out[(long long)(n0 + nn) * K + k0 + tx] = (_Float16)t[tx][nn];
  }
}

// ---- softmax over the QUERY axis (reference quirk): column softmax --------
__global__ __launch_bounds__(256) void col_softmax(float* __restrict__ s, int S, int causal) {
  float* p = s + (long long)blockIdx.y * S * S;
  int k = blockIdx.x * blockDim.x + threadIdx.x;
  if (k >= S) return;
  int q0 = causal ? k : 0;
  float m = -3.4e38f;
  for (int q = q0; q < S; ++q) m = fmaxf(m, p[(long long)q * S + k]);
  float d = 0.f;
  for (int q = q0; q < S; ++q) d += expf(p[(long long)q * S + k] - m);
  float inv = 1.f / d;
  for (int q = 0; q < S; ++q) {
    float v = (q >= q0) ? expf(p[(long long)q * S + k] - m) * inv : 0.f;
    p[(long long)q * S + k] = v;
  }
}

// ---- LayerNorm with fused residual add: out = LN(a + b) -------------------
__global__ __launch_bounds__(256) void layernorm_add(
    const float* __restrict__ a, const float* __restrict__ b,
    const float* __restrict__ g, const float* __restrict__ beta,
    float* __restrict__ o, int Dd) {
  __shared__ float red[256];
  const long long row = blockIdx.x;
  const float* pa = a + row * Dd;
  const float* pb = b ? b + row * Dd : nullptr;
  float* po = o + row * Dd;
  int tid = threadIdx.x;
  float s = 0.f;
  for (int i = tid; i < Dd; i += 256) s += pa[i] + (pb ? pb[i] : 0.f);
  red[tid] = s; __syncthreads();
  for (int w = 128; w > 0; w >>= 1) { if (tid < w) red[tid] += red[tid + w]; __syncthreads(); }
  float mean = red[0] / Dd;
  __syncthreads();
  float s2 = 0.f;
  for (int i = tid; i < Dd; i += 256) { float v = pa[i] + (pb ? pb[i] : 0.f) - mean; s2 += v * v; }
  red[tid] = s2; __syncthreads();
  for (int w = 128; w > 0; w >>= 1) { if (tid < w) red[tid] += red[tid + w]; __syncthreads(); }
  float inv = rsqrtf(red[0] / Dd + 1e-5f);
  for (int i = tid; i < Dd; i += 256) {
    float v = pa[i] + (pb ? pb[i] : 0.f);
    po[i] = (v - mean) * inv * g[i] + beta[i];
  }
}

// ---- embedding lookup + sinusoidal pos-enc (per-index power, ref quirk) ---
__global__ __launch_bounds__(256) void embed_pos(
    const int* __restrict__ toks, const float* __restrict__ emb,
    float* __restrict__ x, int S, int Dd) {
  int bs = blockIdx.x;
  int s  = bs % S;
  long long tok = toks[bs];
  const float* e = emb + tok * Dd;
  float* px = x + (long long)bs * Dd;
  for (int d = threadIdx.x; d < Dd; d += 256) {
    float power = powf(10000.f, 2.f * (float)d / (float)Dd);
    float ang = (float)s / power;
    float pe = ((d & 1) == 0) ? sinf(ang) : cosf(ang);
    px[d] = e[d] + pe;
  }
}

// ---- last-token logits only ----------------------------------------------
__global__ __launch_bounds__(256) void last_logits(
    const float* __restrict__ y, const float* __restrict__ W,
    const float* __restrict__ bias, float* __restrict__ out,
    int S, int Dd, int Vv) {
  int v = blockIdx.x * 256 + threadIdx.x;
  int b = blockIdx.y;
  if (v >= Vv) return;
  const float* yr = y + ((long long)b * S + (S - 1)) * Dd;
  float acc = bias[v];
  for (int k = 0; k < Dd; ++k) acc += yr[k] * W[(long long)k * Vv + v];
  out[(long long)b * Vv + v] = acc;
}

extern "C" void kernel_launch(void* const* d_in, const int* in_sizes, int n_in,
                              void* d_out, int out_size, void* d_ws, size_t ws_size,
                              hipStream_t stream) {
  (void)in_sizes; (void)n_in; (void)out_size; (void)ws_size;
  const int Bb = 2, S = 1024, Dd = 1024, Hh = 16, DHd = 64, FFd = 4096, Vv = 32000, Ll = 4;
  int i = 0;
  const float* emb    = (const float*)d_in[i++];
  const float* eWQ    = (const float*)d_in[i++];
  const float* eWK    = (const float*)d_in[i++];
  const float* eWV    = (const float*)d_in[i++];
  const float* eWO    = (const float*)d_in[i++];
  const float* eln1_g = (const float*)d_in[i++];
  const float* eln1_b = (const float*)d_in[i++];
  const float* eln2_g = (const float*)d_in[i++];
  const float* eln2_b = (const float*)d_in[i++];
  const float* ew1    = (const float*)d_in[i++];
  const float* eb1    = (const float*)d_in[i++];
  const float* ew2    = (const float*)d_in[i++];
  const float* eb2    = (const float*)d_in[i++];
  const float* dWQ1   = (const float*)d_in[i++];
  const float* dWK1   = (const float*)d_in[i++];
  const float* dWV1   = (const float*)d_in[i++];
  const float* dWO1   = (const float*)d_in[i++];
  const float* dWQ2   = (const float*)d_in[i++];
  const float* dWK2   = (const float*)d_in[i++];
  const float* dWV2   = (const float*)d_in[i++];
  const float* dWO2   = (const float*)d_in[i++];
  const float* dln1_g = (const float*)d_in[i++];
  const float* dln1_b = (const float*)d_in[i++];
  const float* dln2_g = (const float*)d_in[i++];
  const float* dln2_b = (const float*)d_in[i++];
  const float* dln3_g = (const float*)d_in[i++];
  const float* dln3_b = (const float*)d_in[i++];
  const float* dw1    = (const float*)d_in[i++];
  const float* db1    = (const float*)d_in[i++];
  const float* dw2    = (const float*)d_in[i++];
  const float* db2    = (const float*)d_in[i++];
  const float* out_w  = (const float*)d_in[i++];
  const float* out_b  = (const float*)d_in[i++];
  const int*   inseq  = (const int*)d_in[i++];
  const int*   outseq = (const int*)d_in[i++];
  float* out = (float*)d_out;

  char* wsb = (char*)d_ws;
  size_t off = 0;
  auto allocF = [&](size_t n) -> float* {
    float* p = (float*)(wsb + off); off += ((n * 4 + 255) / 256) * 256; return p;
  };
  auto allocH = [&](size_t n) -> _Float16* {
    _Float16* p = (_Float16*)(wsb + off); off += ((n * 2 + 255) / 256) * 256; return p;
  };
  const size_t BSD = (size_t)Bb * S * Dd;
  float* x   = allocF(BSD);
  float* y   = allocF(BSD);
  float* t1  = allocF(BSD);
  float* t2  = allocF(BSD);
  float* t3  = allocF(BSD);
  float* att = allocF(BSD);
  float* Qb  = allocF((size_t)Bb * Hh * S * DHd);
  float* Kb  = allocF((size_t)Bb * Hh * S * DHd);
  float* Vb  = allocF((size_t)Bb * Hh * S * DHd);
  float* Sc  = allocF((size_t)Bb * Hh * S * S);
  float* Fh  = allocF((size_t)Bb * S * FFd);
  // f16 operand copies
  _Float16* aq16  = allocH(BSD);
  _Float16* akv16 = allocH(BSD);
  _Float16* Qb16  = allocH((size_t)Bb * Hh * S * DHd);
  _Float16* Kb16  = allocH((size_t)Bb * Hh * S * DHd);
  _Float16* Vt16  = allocH((size_t)Bb * Hh * DHd * S);
  _Float16* Sc16  = allocH((size_t)Bb * Hh * S * S);
  _Float16* att16 = allocH(BSD);
  _Float16* Fh16  = allocH((size_t)Bb * S * FFd);
  // f16 transposed weights [N][K]
  const size_t szP = (size_t)Ll * Hh * Dd * DHd, szO = (size_t)Ll * Dd * Dd, szF = (size_t)Ll * Dd * FFd;
  _Float16* eWQt  = allocH(szP); _Float16* eWKt  = allocH(szP); _Float16* eWVt  = allocH(szP);
  _Float16* eWOt  = allocH(szO); _Float16* ew1t  = allocH(szF); _Float16* ew2t  = allocH(szF);
  _Float16* dWQ1t = allocH(szP); _Float16* dWK1t = allocH(szP); _Float16* dWV1t = allocH(szP);
  _Float16* dWO1t = allocH(szO);
  _Float16* dWQ2t = allocH(szP); _Float16* dWK2t = allocH(szP); _Float16* dWV2t = allocH(szP);
  _Float16* dWO2t = allocH(szO); _Float16* dw1t  = allocH(szF); _Float16* dw2t  = allocH(szF);

  auto cvt = [&](const float* in, _Float16* o16, long long n) {
    cvt_f16_kernel<<<4096, 256, 0, stream>>>(in, o16, n);
  };
  auto tpose = [&](const float* in, _Float16* o16, int K, int N, int batches) {
    tpose_f16_kernel<<<dim3(N / 32, K / 32, batches), 256, 0, stream>>>(
        in, (long long)K * N, N, o16, (long long)K * N, K, N);
  };
  auto gemm = [&](const _Float16* A, long long sAb, long long sAh, int lda,
                  const _Float16* Bt, long long sBb, long long sBh, int ldb,
                  float* C, long long sCb, long long sCh, int ldc,
                  const float* bias, int M, int N, int K, int innerH, int batches,
                  float alpha, int relu) {
    dim3 g(N / BN, M / BM, batches);
    gemm_f16_wmma<<<g, TPB, 0, stream>>>(A, sAb, sAh, lda, Bt, sBb, sBh, ldb,
                                         C, sCb, sCh, ldc, bias, M, N, K, innerH, alpha, relu);
  };

  // ---- one-time weight transpose+convert ----------------------------------
  tpose(eWQ,  eWQt,  Dd, DHd, Ll * Hh); tpose(eWK,  eWKt,  Dd, DHd, Ll * Hh);
  tpose(eWV,  eWVt,  Dd, DHd, Ll * Hh); tpose(eWO,  eWOt,  Dd, Dd,  Ll);
  tpose(ew1,  ew1t,  Dd, FFd, Ll);      tpose(ew2,  ew2t,  FFd, Dd, Ll);
  tpose(dWQ1, dWQ1t, Dd, DHd, Ll * Hh); tpose(dWK1, dWK1t, Dd, DHd, Ll * Hh);
  tpose(dWV1, dWV1t, Dd, DHd, Ll * Hh); tpose(dWO1, dWO1t, Dd, Dd,  Ll);
  tpose(dWQ2, dWQ2t, Dd, DHd, Ll * Hh); tpose(dWK2, dWK2t, Dd, DHd, Ll * Hh);
  tpose(dWV2, dWV2t, Dd, DHd, Ll * Hh); tpose(dWO2, dWO2t, Dd, Dd,  Ll);
  tpose(dw1,  dw1t,  Dd, FFd, Ll);      tpose(dw2,  dw2t,  FFd, Dd, Ll);

  const long long sX  = (long long)S * Dd;
  const long long sQh = (long long)S * DHd;
  const long long sQb = (long long)Hh * S * DHd;
  const long long sSh = (long long)S * S;
  const long long sSb = (long long)Hh * S * S;
  const long long sW  = (long long)Dd * DHd;
  const float atnscale = 1.0f / 32.0f;            // 1/sqrt(Sk=1024), ref quirk

  auto mha = [&](const float* qin, const float* kvin,
                 const _Float16* WQt, const _Float16* WKt, const _Float16* WVt, const _Float16* WOt,
                 float* outp, int causal) {
    cvt(qin, aq16, (long long)BSD);
    const _Float16* kv16 = aq16;
    if (kvin != qin) { cvt(kvin, akv16, (long long)BSD); kv16 = akv16; }
    gemm(aq16, sX, 0, Dd, WQt, 0, sW, Dd, Qb, sQb, sQh, DHd, nullptr, S, DHd, Dd, Hh, Bb * Hh, 1.f, 0);
    gemm(kv16, sX, 0, Dd, WKt, 0, sW, Dd, Kb, sQb, sQh, DHd, nullptr, S, DHd, Dd, Hh, Bb * Hh, 1.f, 0);
    gemm(kv16, sX, 0, Dd, WVt, 0, sW, Dd, Vb, sQb, sQh, DHd, nullptr, S, DHd, Dd, Hh, Bb * Hh, 1.f, 0);
    cvt(Qb, Qb16, (long long)Bb * Hh * S * DHd);
    cvt(Kb, Kb16, (long long)Bb * Hh * S * DHd);   // K_head is [S][DH] == Bt layout already
    gemm(Qb16, sQb, sQh, DHd, Kb16, sQb, sQh, DHd, Sc, sSb, sSh, S,
         nullptr, S, S, DHd, Hh, Bb * Hh, atnscale, 0);
    col_softmax<<<dim3(S / 256, Bb * Hh), 256, 0, stream>>>(Sc, S, causal);
    cvt(Sc, Sc16, (long long)Bb * Hh * S * S);
    tpose(Vb, Vt16, S, DHd, Bb * Hh);              // V_head [S][DH] -> [DH][S]
    gemm(Sc16, sSb, sSh, S, Vt16, sQb, sQh, S, att, sX, (long long)DHd, Dd,
         nullptr, S, DHd, S, Hh, Bb * Hh, 1.f, 0);
    cvt(att, att16, (long long)BSD);
    gemm(att16, 0, 0, Dd, WOt, 0, 0, Dd, outp, 0, 0, Dd, nullptr, Bb * S, Dd, Dd, 1, 1, 1.f, 0);
  };

  // ---------------- encoder ----------------
  embed_pos<<<Bb * S, 256, 0, stream>>>(inseq, emb, x, S, Dd);
  for (int l = 0; l < Ll; ++l) {
    mha(x, x, eWQt + (size_t)l * Hh * Dd * DHd, eWKt + (size_t)l * Hh * Dd * DHd,
        eWVt + (size_t)l * Hh * Dd * DHd, eWOt + (size_t)l * Dd * Dd, t1, 0);
    layernorm_add<<<Bb * S, 256, 0, stream>>>(t1, x, eln1_g + l * Dd, eln1_b + l * Dd, t2, Dd);
    cvt(t2, aq16, (long long)BSD);
    gemm(aq16, 0, 0, Dd, ew1t + (size_t)l * Dd * FFd, 0, 0, Dd,
         Fh, 0, 0, FFd, eb1 + (long long)l * FFd, Bb * S, FFd, Dd, 1, 1, 1.f, 1);
    cvt(Fh, Fh16, (long long)Bb * S * FFd);
    gemm(Fh16, 0, 0, FFd, ew2t + (size_t)l * Dd * FFd, 0, 0, FFd,
         t1, 0, 0, Dd, eb2 + (long long)l * Dd, Bb * S, Dd, FFd, 1, 1, 1.f, 0);
    layernorm_add<<<Bb * S, 256, 0, stream>>>(t1, t2, eln2_g + l * Dd, eln2_b + l * Dd, x, Dd);
  }
  // ---------------- decoder ----------------
  embed_pos<<<Bb * S, 256, 0, stream>>>(outseq, emb, y, S, Dd);
  for (int l = 0; l < Ll; ++l) {
    mha(y, y, dWQ1t + (size_t)l * Hh * Dd * DHd, dWK1t + (size_t)l * Hh * Dd * DHd,
        dWV1t + (size_t)l * Hh * Dd * DHd, dWO1t + (size_t)l * Dd * Dd, t1, 1);
    layernorm_add<<<Bb * S, 256, 0, stream>>>(t1, y, dln1_g + l * Dd, dln1_b + l * Dd, t2, Dd);  // z
    mha(t2, x, dWQ2t + (size_t)l * Hh * Dd * DHd, dWK2t + (size_t)l * Hh * Dd * DHd,
        dWV2t + (size_t)l * Hh * Dd * DHd, dWO2t + (size_t)l * Dd * Dd, t1, 0);
    layernorm_add<<<Bb * S, 256, 0, stream>>>(t1, t2, dln2_g + l * Dd, dln2_b + l * Dd, t3, Dd); // c
    cvt(t3, aq16, (long long)BSD);
    gemm(aq16, 0, 0, Dd, dw1t + (size_t)l * Dd * FFd, 0, 0, Dd,
         Fh, 0, 0, FFd, db1 + (long long)l * FFd, Bb * S, FFd, Dd, 1, 1, 1.f, 1);
    cvt(Fh, Fh16, (long long)Bb * S * FFd);
    gemm(Fh16, 0, 0, FFd, dw2t + (size_t)l * Dd * FFd, 0, 0, FFd,
         t1, 0, 0, Dd, db2 + (long long)l * Dd, Bb * S, Dd, FFd, 1, 1, 1.f, 0);
    layernorm_add<<<Bb * S, 256, 0, stream>>>(t1, t3, dln3_g + l * Dd, dln3_b + l * Dd, y, Dd);  // y
  }
  last_logits<<<dim3((Vv + 255) / 256, Bb), 256, 0, stream>>>(y, out_w, out_b, out, S, Dd, Vv);
}
